// EdgeNet_edge_old_45526653337877
// MI455X (gfx1250) — compile-verified, matrix operand order checked by reading, not run on hardware
//
#include <hip/hip_runtime.h>
#include <stdint.h>

// ---------------- problem constants ----------------
constexpr int NN = 100000;
constexpr int EE = 1600000;
constexpr int SLOTS = 256;

// ---------------- workspace layout (float indices into ws) ----------------
// [0, 6912) as uint32: 27 weight tiles * 256 u32 (f16 B-fragments)
constexpr int AX = 8192, BX = 8195;        // bn_in  scale/shift (3)
constexpr int AZ = 8198, BZ = 8200;        // bn_lat scale/shift (2)
constexpr int AXC = 8202, BXC = 8206;      // ep_bn1 scale/shift (4)
constexpr int AH = 8210, BH = 8242;        // ep_bn2 scale/shift (32)
constexpr int AR = 8274, BR = 8306;        // ep_bn3 scale/shift (32)
constexpr int ZERO_BASE = 8448;
constexpr int Z_SUM = ZERO_BASE;           // N*2
constexpr int CNT   = Z_SUM + 2 * NN;      // N  (tgt in-degree, float)
constexpr int DEG   = CNT + NN;            // N  (src out-degree, float)
constexpr int XR    = DEG + NN;            // N*3
constexpr int SX_S  = XR + 3 * NN;         // SLOTS*3
constexpr int SX_Q  = SX_S + SLOTS * 3;
constexpr int SZ_S  = SX_Q + SLOTS * 3;    // SLOTS*2
constexpr int SZ_Q  = SZ_S + SLOTS * 2;
constexpr int SXC_S = SZ_Q + SLOTS * 2;    // SLOTS*4
constexpr int SXC_Q = SXC_S + SLOTS * 4;
constexpr int SH_S  = SXC_Q + SLOTS * 4;   // SLOTS*32
constexpr int SH_Q  = SH_S + SLOTS * 32;
constexpr int SR_S  = SH_Q + SLOTS * 32;   // SLOTS*32
constexpr int SR_Q  = SR_S + SLOTS * 32;
constexpr int ZERO_END = SR_Q + SLOTS * 32;
constexpr int Z_BN  = ZERO_END;            // N*2
constexpr int ZCOUNT = ZERO_END - ZERO_BASE;

// ---------------- weight tile indices ----------------
constexpr int T_ENC1 = 0, T_ENC2 = 2, T_ENC3 = 6, T_ENC4 = 10;
constexpr int T_DEC1 = 11, T_DEC2 = 13, T_DEC3 = 17, T_DEC4 = 21;
constexpr int T_FC1 = 22, T_FC2 = 24, T_FC4 = 26;
constexpr int NTILES = 27;

// ---------------- WMMA types ----------------
typedef __attribute__((ext_vector_type(16))) _Float16 v16h;
typedef __attribute__((ext_vector_type(8)))  float    v8f;

union Frag {
  v16h h;
  uint32_t u[8];
  _Float16 f[16];
};

__device__ __forceinline__ uint32_t pkh(float a, float b) {
  union { _Float16 h[2]; uint32_t u; } t;
  t.h[0] = (_Float16)a; t.h[1] = (_Float16)b;
  return t.u;
}

__device__ __forceinline__ v8f splat8(float x) {
  v8f c;
#pragma unroll
  for (int i = 0; i < 8; ++i) c[i] = x;
  return c;
}

// single-instruction relu (avoids fmaxf's canonicalize; max_num quiets NaN)
__device__ __forceinline__ float relu1(float x) {
  float y;
  asm("v_max_num_f32 %0, 0, %1" : "=v"(y) : "v"(x));
  return y;
}

__device__ __forceinline__ v8f wmma16(const Frag& a, const Frag& b, v8f c) {
  return __builtin_amdgcn_wmma_f32_16x16x32_f16(
      false, a.h, false, b.h, (short)0, c, false, false);
}

__device__ __forceinline__ Frag load_b(const uint32_t* wf, int tile, int lane) {
  Frag b;
  const uint32_t* p = wf + tile * 256 + lane;
#pragma unroll
  for (int v = 0; v < 8; ++v) b.u[v] = p[v * 32];
  return b;
}

// intra-wave staging fence: per-wave LDS ops are processed in order on CDNA5,
// we only need to stop compiler reordering of the store->load sequence.
__device__ __forceinline__ void wave_fence() { __builtin_amdgcn_wave_barrier(); }

// LDS stride in halves for a 16-row activation tile
constexpr int ST = 66;

// A-fragment (16x32 f16) load from an LDS tile [16][ST] at column base kbase.
__device__ __forceinline__ Frag load_a_lds(const _Float16* wlds, int lane, int kbase) {
  Frag a;
  const int r = lane & 15, sel = lane >> 4;
  const _Float16* row = wlds + r * ST + kbase;
#pragma unroll
  for (int v = 0; v < 8; ++v) {
    const int k = ((v & 3) << 1) + (sel << 3) + ((v >> 2) << 4);
    a.f[2 * v]     = row[k];
    a.f[2 * v + 1] = row[k + 1];
  }
  return a;
}

// Store a 16x16 f32 D tile (bias already folded) to LDS as f16, relu.
__device__ __forceinline__ void stage_relu(v8f c, _Float16* wlds, int lane, int nt) {
  const int n = (lane & 15) + nt * 16;
  const int m0 = (lane >> 4) * 8;
#pragma unroll
  for (int v = 0; v < 8; ++v)
    wlds[(m0 + v) * ST + n] = (_Float16)relu1(c[v]);
}

// relu then per-channel affine (folds a BatchNorm into the staging).
__device__ __forceinline__ void stage_relu_affine(v8f c, _Float16* wlds, int lane, int nt,
                                                  const float* scale, const float* shift) {
  const int n = (lane & 15) + nt * 16;
  const int m0 = (lane >> 4) * 8;
  const float sc = scale[n], sh = shift[n];
#pragma unroll
  for (int v = 0; v < 8; ++v)
    wlds[(m0 + v) * ST + n] = (_Float16)fmaf(relu1(c[v]), sc, sh);
}

// ---------------- parameter structs ----------------
struct P {
  const float* x; const int* ei; const float* ea;
  const float* enc_b1; const float* enc_b2; const float* enc_b3;
  const float* dec_b1; const float* dec_b2; const float* dec_b3;
  const float* fc2_b; const float* fc4_b;
  float* ws; float* out;
};
struct PW { const float* W[11]; const float* Wb[11]; };

struct TD { int widx, in_dim, out_dim, k0, n0; };
__constant__ TD g_tiles[NTILES] = {
  {0,6,32,0,0},{0,6,32,0,16},
  {1,32,64,0,0},{1,32,64,0,16},{1,32,64,0,32},{1,32,64,0,48},
  {2,64,32,0,0},{2,64,32,32,0},{2,64,32,0,16},{2,64,32,32,16},
  {3,32,2,0,0},
  {4,4,32,0,0},{4,4,32,0,16},
  {5,32,64,0,0},{5,32,64,0,16},{5,32,64,0,32},{5,32,64,0,48},
  {6,64,32,0,0},{6,64,32,32,0},{6,64,32,0,16},{6,64,32,32,16},
  {7,32,3,0,0},
  {8,4,32,0,0},{8,4,32,0,16},
  {9,32,32,0,0},{9,32,32,0,16},
  {10,32,3,0,0}
};

// ---------------- utility kernels ----------------
__global__ void k_zero(float* ws) {
  int i = blockIdx.x * 256 + threadIdx.x;
  if (i < ZCOUNT) ws[ZERO_BASE + i] = 0.f;
}

// Build f16 B-fragments. B[k][n] = W[n][k]. For padded-K layers (in_dim < 32)
// the bias is folded in as a constant-1 input channel at k == in_dim.
__global__ void k_prep(PW pw, uint32_t* wf) {
  const int lane = threadIdx.x & 31;
  const int v = threadIdx.x >> 5;
  const int sel = lane >> 4, n = lane & 15;
  const int kp = ((v & 3) << 1) + (sel << 3) + ((v >> 2) << 4);
  for (int t = 0; t < NTILES; ++t) {
    TD d = g_tiles[t];
    const float* W = pw.W[d.widx];
    float f0 = 0.f, f1 = 0.f;
    const int nn = d.n0 + n;
    if (nn < d.out_dim) {
      const int k0 = d.k0 + kp;
      if (k0 < d.in_dim)     f0 = W[nn * d.in_dim + k0];
      if (k0 + 1 < d.in_dim) f1 = W[nn * d.in_dim + k0 + 1];
      if (d.in_dim < 32 && k0 == d.in_dim) f0 = pw.Wb[d.widx][nn];  // bias row
    }
    wf[t * 256 + v * 32 + lane] = pkh(f0, f1);
  }
}

// per-channel mean/var stats of x over N nodes (3 channels), slot atomics
__global__ void k_xstats(P p) {
  __shared__ float ls[3], lq[3];
  if (threadIdx.x < 3) { ls[threadIdx.x] = 0.f; lq[threadIdx.x] = 0.f; }
  __syncthreads();
  int i = blockIdx.x * 256 + threadIdx.x;
  if (i < NN) {
#pragma unroll
    for (int c = 0; c < 3; ++c) {
      float v = p.x[3 * i + c];
      atomicAdd(&ls[c], v);
      atomicAdd(&lq[c], v * v);
    }
  }
  __syncthreads();
  int slot = blockIdx.x & (SLOTS - 1);
  if (threadIdx.x < 3) {
    atomicAdd(&p.ws[SX_S + slot * 3 + threadIdx.x], ls[threadIdx.x]);
    atomicAdd(&p.ws[SX_Q + slot * 3 + threadIdx.x], lq[threadIdx.x]);
  }
}

// fold BN stats into scale/shift: a = g/sqrt(var+eps), b = beta - mean*a
__global__ void k_finalize(const float* ssum, const float* ssq,
                           const float* g, const float* beta,
                           float* a_out, float* b_out, int C, float denom) {
  int c = threadIdx.x;
  if (c >= C) return;
  float s = 0.f, q = 0.f;
  for (int k = 0; k < SLOTS; ++k) { s += ssum[k * C + c]; q += ssq[k * C + c]; }
  float m = s / denom;
  float v = q / denom - m * m;
  float a = g[c] * rsqrtf(v + 1e-5f);
  a_out[c] = a;
  b_out[c] = beta[c] - m * a;
}

// ---------------- encoder: per-edge MLP 6->32->64->32->2 + z segment-sum ----------------
__global__ __launch_bounds__(256) void k_encoder(P p) {
  __shared__ _Float16 sA[8][16 * ST];
  __shared__ _Float16 sB[8][16 * ST];
  const int lane = threadIdx.x & 31;
  const int wave = threadIdx.x >> 5;
  const int sel = lane >> 4, r = lane & 15;
  const uint32_t* wf = (const uint32_t*)p.ws;
  float* ws = p.ws;
  const int* srcI = p.ei;
  const int* tgtI = p.ei + EE;
  const int edge0 = (blockIdx.x * 8 + wave) * 16;

  // layer-1 A fragment: [x_bn[tgt] (3), edge_attr (3), 1.0 bias channel at k=6]
  Frag a;
#pragma unroll
  for (int i = 0; i < 8; ++i) a.u[i] = 0;
  if (!sel) {
    const int e = edge0 + r;
    const int t = tgtI[e];
    float f0 = p.x[t * 3 + 0] * ws[AX + 0] + ws[BX + 0];
    float f1 = p.x[t * 3 + 1] * ws[AX + 1] + ws[BX + 1];
    float f2 = p.x[t * 3 + 2] * ws[AX + 2] + ws[BX + 2];
    a.u[0] = pkh(f0, f1);
    a.u[1] = pkh(f2, p.ea[e * 3 + 0]);
    a.u[2] = pkh(p.ea[e * 3 + 1], p.ea[e * 3 + 2]);
    a.u[3] = pkh(1.f, 0.f);              // bias channel (k=6)
  }
  _Float16* LA = sA[wave];
  _Float16* LB = sB[wave];
  const int nlo = lane & 15;
  // L1: K=32 -> 32 (bias folded into weights, C = inline 0)
  {
    Frag b0 = load_b(wf, T_ENC1 + 0, lane);
    Frag b1 = load_b(wf, T_ENC1 + 1, lane);
    v8f c0 = {}; c0 = wmma16(a, b0, c0);
    v8f c1 = {}; c1 = wmma16(a, b1, c1);
    stage_relu(c0, LA, lane, 0);
    stage_relu(c1, LA, lane, 1);
  }
  wave_fence();
  // L2: 32 -> 64
  {
    Frag a2 = load_a_lds(LA, lane, 0);
    v8f acc[4];
#pragma unroll
    for (int nt = 0; nt < 4; ++nt) {
      Frag b = load_b(wf, T_ENC2 + nt, lane);
      acc[nt] = splat8(p.enc_b1[nt * 16 + nlo]);
      acc[nt] = wmma16(a2, b, acc[nt]);
    }
#pragma unroll
    for (int nt = 0; nt < 4; ++nt) stage_relu(acc[nt], LB, lane, nt);
  }
  wave_fence();
  // L3: 64 -> 32 (two K tiles)
  {
    Frag a0 = load_a_lds(LB, lane, 0);
    Frag a1 = load_a_lds(LB, lane, 32);
    v8f acc[2];
#pragma unroll
    for (int nt = 0; nt < 2; ++nt) {
      acc[nt] = splat8(p.enc_b2[nt * 16 + nlo]);
      Frag b = load_b(wf, T_ENC3 + nt * 2 + 0, lane); acc[nt] = wmma16(a0, b, acc[nt]);
      b = load_b(wf, T_ENC3 + nt * 2 + 1, lane);      acc[nt] = wmma16(a1, b, acc[nt]);
    }
#pragma unroll
    for (int nt = 0; nt < 2; ++nt) stage_relu(acc[nt], LA, lane, nt);
  }
  wave_fence();
  // L4: 32 -> 2 (relu), atomic z segment-sum + degree counts
  {
    Frag a4 = load_a_lds(LA, lane, 0);
    Frag b = load_b(wf, T_ENC4, lane);
    v8f c = splat8(nlo < 2 ? p.enc_b3[nlo] : 0.f);
    c = wmma16(a4, b, c);
    if (nlo < 2) {
#pragma unroll
      for (int v = 0; v < 8; ++v) {
        const int e2 = edge0 + v + 8 * sel;
        const int t2 = tgtI[e2];
        atomicAdd(&ws[Z_SUM + t2 * 2 + nlo], relu1(c[v]));
      }
    }
    if (!sel) {
      const int e2 = edge0 + r;
      atomicAdd(&ws[CNT + tgtI[e2]], 1.f);
      atomicAdd(&ws[DEG + srcI[e2]], 1.f);
    }
  }
}

// z = z_sum / max(cnt,1); accumulate BN(z) stats
__global__ void k_znode(P p) {
  __shared__ float ls[2], lq[2];
  if (threadIdx.x < 2) { ls[threadIdx.x] = 0.f; lq[threadIdx.x] = 0.f; }
  __syncthreads();
  int i = blockIdx.x * 256 + threadIdx.x;
  if (i < NN) {
    float c = fmaxf(p.ws[CNT + i], 1.f);
    float z0 = p.ws[Z_SUM + 2 * i] / c;
    float z1 = p.ws[Z_SUM + 2 * i + 1] / c;
    p.ws[Z_BN + 2 * i] = z0;
    p.ws[Z_BN + 2 * i + 1] = z1;
    atomicAdd(&ls[0], z0); atomicAdd(&lq[0], z0 * z0);
    atomicAdd(&ls[1], z1); atomicAdd(&lq[1], z1 * z1);
  }
  __syncthreads();
  int slot = blockIdx.x & (SLOTS - 1);
  if (threadIdx.x < 2) {
    atomicAdd(&p.ws[SZ_S + slot * 2 + threadIdx.x], ls[threadIdx.x]);
    atomicAdd(&p.ws[SZ_Q + slot * 2 + threadIdx.x], lq[threadIdx.x]);
  }
}

// z_bn = BN(z); degree-weighted ep_bn1 stats (xc = [z_bn[src], z_bn[tgt]])
__global__ void k_zbn(P p) {
  __shared__ float ls[4], lq[4];
  if (threadIdx.x < 4) { ls[threadIdx.x] = 0.f; lq[threadIdx.x] = 0.f; }
  __syncthreads();
  int i = blockIdx.x * 256 + threadIdx.x;
  if (i < NN) {
    float z0 = p.ws[Z_BN + 2 * i]     * p.ws[AZ]     + p.ws[BZ];
    float z1 = p.ws[Z_BN + 2 * i + 1] * p.ws[AZ + 1] + p.ws[BZ + 1];
    p.ws[Z_BN + 2 * i] = z0;
    p.ws[Z_BN + 2 * i + 1] = z1;
    float ds = p.ws[DEG + i], dt = p.ws[CNT + i];
    atomicAdd(&ls[0], z0 * ds); atomicAdd(&lq[0], z0 * z0 * ds);
    atomicAdd(&ls[1], z1 * ds); atomicAdd(&lq[1], z1 * z1 * ds);
    atomicAdd(&ls[2], z0 * dt); atomicAdd(&lq[2], z0 * z0 * dt);
    atomicAdd(&ls[3], z1 * dt); atomicAdd(&lq[3], z1 * z1 * dt);
  }
  __syncthreads();
  int slot = blockIdx.x & (SLOTS - 1);
  if (threadIdx.x < 4) {
    atomicAdd(&p.ws[SXC_S + slot * 4 + threadIdx.x], ls[threadIdx.x]);
    atomicAdd(&p.ws[SXC_Q + slot * 4 + threadIdx.x], lq[threadIdx.x]);
  }
}

// fc1 input fragment (xc normalized by ep_bn1, 1.0 bias channel at k=4)
__device__ __forceinline__ Frag make_xc_frag(const float* ws, int sel,
                                             float zj0, float zj1, float zi0, float zi1) {
  Frag ax;
#pragma unroll
  for (int i = 0; i < 8; ++i) ax.u[i] = 0;
  if (!sel) {
    float f0 = zj0 * ws[AXC + 0] + ws[BXC + 0];
    float f1 = zj1 * ws[AXC + 1] + ws[BXC + 1];
    float f2 = zi0 * ws[AXC + 2] + ws[BXC + 2];
    float f3 = zi1 * ws[AXC + 3] + ws[BXC + 3];
    ax.u[0] = pkh(f0, f1); ax.u[1] = pkh(f2, f3);
    ax.u[2] = pkh(1.f, 0.f);             // bias channel (k=4)
  }
  return ax;
}

// ---------------- decoder MLP + fc1 (ep_bn2 stats) ----------------
__global__ __launch_bounds__(256) void k_dec(P p) {
  __shared__ _Float16 sA[8][16 * ST];
  __shared__ _Float16 sB[8][16 * ST];
  __shared__ float hs[32], hq[32];
  const int lane = threadIdx.x & 31;
  const int wave = threadIdx.x >> 5;
  const int sel = lane >> 4, r = lane & 15;
  const uint32_t* wf = (const uint32_t*)p.ws;
  float* ws = p.ws;
  const int* srcI = p.ei;
  const int* tgtI = p.ei + EE;
  const int edge0 = (blockIdx.x * 8 + wave) * 16;
  if (threadIdx.x < 32) { hs[threadIdx.x] = 0.f; hq[threadIdx.x] = 0.f; }
  __syncthreads();

  float zi0 = 0.f, zi1 = 0.f, zj0 = 0.f, zj1 = 0.f;
  if (!sel) {
    const int e = edge0 + r;
    const int s = srcI[e], t = tgtI[e];
    zi0 = ws[Z_BN + 2 * t]; zi1 = ws[Z_BN + 2 * t + 1];
    zj0 = ws[Z_BN + 2 * s]; zj1 = ws[Z_BN + 2 * s + 1];
  }

  // decoder A: [zi, zj - zi, 1.0 bias channel at k=4]
  Frag a;
#pragma unroll
  for (int i = 0; i < 8; ++i) a.u[i] = 0;
  if (!sel) {
    a.u[0] = pkh(zi0, zi1);
    a.u[1] = pkh(zj0 - zi0, zj1 - zi1);
    a.u[2] = pkh(1.f, 0.f);              // bias channel (k=4)
  }

  _Float16* LA = sA[wave];
  _Float16* LB = sB[wave];
  const int nlo = lane & 15;
  // L1 (bias in weights, C = inline 0)
  {
    Frag b0 = load_b(wf, T_DEC1 + 0, lane);
    Frag b1 = load_b(wf, T_DEC1 + 1, lane);
    v8f c0 = {}; c0 = wmma16(a, b0, c0);
    v8f c1 = {}; c1 = wmma16(a, b1, c1);
    stage_relu(c0, LA, lane, 0);
    stage_relu(c1, LA, lane, 1);
  }
  wave_fence();
  // L2
  {
    Frag a2 = load_a_lds(LA, lane, 0);
    v8f acc[4];
#pragma unroll
    for (int nt = 0; nt < 4; ++nt) {
      Frag b = load_b(wf, T_DEC2 + nt, lane);
      acc[nt] = splat8(p.dec_b1[nt * 16 + nlo]);
      acc[nt] = wmma16(a2, b, acc[nt]);
    }
#pragma unroll
    for (int nt = 0; nt < 4; ++nt) stage_relu(acc[nt], LB, lane, nt);
  }
  wave_fence();
  // L3
  {
    Frag a0 = load_a_lds(LB, lane, 0);
    Frag a1 = load_a_lds(LB, lane, 32);
    v8f acc[2];
#pragma unroll
    for (int nt = 0; nt < 2; ++nt) {
      acc[nt] = splat8(p.dec_b2[nt * 16 + nlo]);
      Frag b = load_b(wf, T_DEC3 + nt * 2 + 0, lane); acc[nt] = wmma16(a0, b, acc[nt]);
      b = load_b(wf, T_DEC3 + nt * 2 + 1, lane);      acc[nt] = wmma16(a1, b, acc[nt]);
    }
#pragma unroll
    for (int nt = 0; nt < 2; ++nt) stage_relu(acc[nt], LA, lane, nt);
  }
  wave_fence();
  // L4: 32 -> 3, no relu, atomic x_recon segment-sum
  {
    Frag a4 = load_a_lds(LA, lane, 0);
    Frag b = load_b(wf, T_DEC4, lane);
    v8f c = splat8(nlo < 3 ? p.dec_b3[nlo] : 0.f);
    c = wmma16(a4, b, c);
    if (nlo < 3) {
#pragma unroll
      for (int v = 0; v < 8; ++v) {
        const int e2 = edge0 + v + 8 * sel;
        const int t2 = tgtI[e2];
        atomicAdd(&ws[XR + t2 * 3 + nlo], c[v]);
      }
    }
  }
  // fc1: xc -> 32 (bias in weights), relu; ep_bn2 stats from D registers
  {
    Frag ax = make_xc_frag(ws, sel, zj0, zj1, zi0, zi1);
    Frag b0 = load_b(wf, T_FC1 + 0, lane);
    Frag b1 = load_b(wf, T_FC1 + 1, lane);
    v8f c0 = {}; c0 = wmma16(ax, b0, c0);
    v8f c1 = {}; c1 = wmma16(ax, b1, c1);
#pragma unroll
    for (int nt = 0; nt < 2; ++nt) {
      v8f c = nt ? c1 : c0;
      float s1 = 0.f, s2 = 0.f;
#pragma unroll
      for (int v = 0; v < 8; ++v) {
        float h = relu1(c[v]);
        s1 += h; s2 += h * h;
      }
      atomicAdd(&hs[nt * 16 + nlo], s1);
      atomicAdd(&hq[nt * 16 + nlo], s2);
    }
  }
  __syncthreads();
  int slot = blockIdx.x & (SLOTS - 1);
  if (threadIdx.x < 32) {
    atomicAdd(&ws[SH_S + slot * 32 + threadIdx.x], hs[threadIdx.x]);
    atomicAdd(&ws[SH_Q + slot * 32 + threadIdx.x], hq[threadIdx.x]);
  }
}

// ---------------- h_res pass: stats of relu(BN(h) @ fc2 + b2) ----------------
__global__ __launch_bounds__(256) void k_hres(P p) {
  __shared__ _Float16 sA[8][16 * ST];
  __shared__ float rs[32], rq[32];
  const int lane = threadIdx.x & 31;
  const int wave = threadIdx.x >> 5;
  const int sel = lane >> 4, r = lane & 15;
  const uint32_t* wf = (const uint32_t*)p.ws;
  float* ws = p.ws;
  const int* srcI = p.ei;
  const int* tgtI = p.ei + EE;
  const int edge0 = (blockIdx.x * 8 + wave) * 16;
  if (threadIdx.x < 32) { rs[threadIdx.x] = 0.f; rq[threadIdx.x] = 0.f; }
  __syncthreads();

  float zi0 = 0.f, zi1 = 0.f, zj0 = 0.f, zj1 = 0.f;
  if (!sel) {
    const int e = edge0 + r;
    const int s = srcI[e], t = tgtI[e];
    zi0 = ws[Z_BN + 2 * t]; zi1 = ws[Z_BN + 2 * t + 1];
    zj0 = ws[Z_BN + 2 * s]; zj1 = ws[Z_BN + 2 * s + 1];
  }
  _Float16* LA = sA[wave];
  const int nlo = lane & 15;
  // fc1 -> relu -> ep_bn2 affine, staged f16
  {
    Frag ax = make_xc_frag(ws, sel, zj0, zj1, zi0, zi1);
    Frag b0 = load_b(wf, T_FC1 + 0, lane);
    Frag b1 = load_b(wf, T_FC1 + 1, lane);
    v8f c0 = {}; c0 = wmma16(ax, b0, c0);
    v8f c1 = {}; c1 = wmma16(ax, b1, c1);
    stage_relu_affine(c0, LA, lane, 0, ws + AH, ws + BH);
    stage_relu_affine(c1, LA, lane, 1, ws + AH, ws + BH);
  }
  wave_fence();
  // fc2; accumulate relu(h_res) stats
  {
    Frag ah = load_a_lds(LA, lane, 0);
    v8f acc[2];
#pragma unroll
    for (int nt = 0; nt < 2; ++nt) {
      Frag b = load_b(wf, T_FC2 + nt, lane);
      acc[nt] = splat8(p.fc2_b[nt * 16 + nlo]);
      acc[nt] = wmma16(ah, b, acc[nt]);
    }
#pragma unroll
    for (int nt = 0; nt < 2; ++nt) {
      float s1 = 0.f, s2 = 0.f;
#pragma unroll
      for (int v = 0; v < 8; ++v) {
        float hv = relu1(acc[nt][v]);
        s1 += hv; s2 += hv * hv;
      }
      atomicAdd(&rs[nt * 16 + nlo], s1);
      atomicAdd(&rq[nt * 16 + nlo], s2);
    }
  }
  __syncthreads();
  int slot = blockIdx.x & (SLOTS - 1);
  if (threadIdx.x < 32) {
    atomicAdd(&ws[SR_S + slot * 32 + threadIdx.x], rs[threadIdx.x]);
    atomicAdd(&ws[SR_Q + slot * 32 + threadIdx.x], rq[threadIdx.x]);
  }
}

// ---------------- final pred pass ----------------
__global__ __launch_bounds__(256) void k_pred(P p) {
  __shared__ _Float16 sA[8][16 * ST];
  __shared__ _Float16 sB[8][16 * ST];
  const int lane = threadIdx.x & 31;
  const int wave = threadIdx.x >> 5;
  const int sel = lane >> 4, r = lane & 15;
  const uint32_t* wf = (const uint32_t*)p.ws;
  float* ws = p.ws;
  const int* srcI = p.ei;
  const int* tgtI = p.ei + EE;
  const int edge0 = (blockIdx.x * 8 + wave) * 16;

  float zi0 = 0.f, zi1 = 0.f, zj0 = 0.f, zj1 = 0.f;
  if (!sel) {
    const int e = edge0 + r;
    const int s = srcI[e], t = tgtI[e];
    zi0 = ws[Z_BN + 2 * t]; zi1 = ws[Z_BN + 2 * t + 1];
    zj0 = ws[Z_BN + 2 * s]; zj1 = ws[Z_BN + 2 * s + 1];
  }
  _Float16* LA = sA[wave];
  _Float16* LB = sB[wave];
  const int nlo = lane & 15;
  // fc1 -> h_bn staged
  {
    Frag ax = make_xc_frag(ws, sel, zj0, zj1, zi0, zi1);
    Frag b0 = load_b(wf, T_FC1 + 0, lane);
    Frag b1 = load_b(wf, T_FC1 + 1, lane);
    v8f c0 = {}; c0 = wmma16(ax, b0, c0);
    v8f c1 = {}; c1 = wmma16(ax, b1, c1);
    stage_relu_affine(c0, LA, lane, 0, ws + AH, ws + BH);
    stage_relu_affine(c1, LA, lane, 1, ws + AH, ws + BH);
  }
  wave_fence();
  // fc2 -> h_final = h_bn + ep_bn3(relu(h_res)), staged to LB
  {
    Frag ah = load_a_lds(LA, lane, 0);
    v8f acc[2];
#pragma unroll
    for (int nt = 0; nt < 2; ++nt) {
      Frag b = load_b(wf, T_FC2 + nt, lane);
      acc[nt] = splat8(p.fc2_b[nt * 16 + nlo]);
      acc[nt] = wmma16(ah, b, acc[nt]);
    }
#pragma unroll
    for (int nt = 0; nt < 2; ++nt) {
      const int nn = nt * 16 + nlo;
      const float ar = ws[AR + nn], br = ws[BR + nn];
      const int m0 = sel * 8;
#pragma unroll
      for (int v = 0; v < 8; ++v) {
        float hr = fmaf(relu1(acc[nt][v]), ar, br);
        float hb = (float)LA[(m0 + v) * ST + nn];
        LB[(m0 + v) * ST + nn] = (_Float16)(hb + hr);
      }
    }
  }
  wave_fence();
  // fc4 -> pred_edge_attr
  {
    Frag af = load_a_lds(LB, lane, 0);
    Frag b = load_b(wf, T_FC4, lane);
    v8f c = splat8(nlo < 3 ? p.fc4_b[nlo] : 0.f);
    c = wmma16(af, b, c);
    if (nlo < 3) {
#pragma unroll
      for (int v = 0; v < 8; ++v) {
        const int e2 = edge0 + v + 8 * sel;
        p.out[3 * NN + e2 * 3 + nlo] = c[v];
      }
    }
  }
}

// x_recon = xr_sum / max(cnt,1)
__global__ void k_xrecon(P p) {
  int i = blockIdx.x * 256 + threadIdx.x;
  if (i < NN) {
    float c = fmaxf(p.ws[CNT + i], 1.f);
#pragma unroll
    for (int k = 0; k < 3; ++k) p.out[3 * i + k] = p.ws[XR + 3 * i + k] / c;
  }
}

// ---------------- host launch ----------------
extern "C" void kernel_launch(void* const* d_in, const int* in_sizes, int n_in,
                              void* d_out, int out_size, void* d_ws, size_t ws_size,
                              hipStream_t stream) {
  const float* x  = (const float*)d_in[0];
  const int*   ei = (const int*)d_in[1];
  const float* ea = (const float*)d_in[2];
  float* ws = (float*)d_ws;

  P p;
  p.x = x; p.ei = ei; p.ea = ea;
  p.enc_b1 = (const float*)d_in[6];
  p.enc_b2 = (const float*)d_in[8];
  p.enc_b3 = (const float*)d_in[10];
  p.dec_b1 = (const float*)d_in[14];
  p.dec_b2 = (const float*)d_in[16];
  p.dec_b3 = (const float*)d_in[18];
  p.fc2_b  = (const float*)d_in[22];
  p.fc4_b  = (const float*)d_in[24];
  p.ws = ws;
  p.out = (float*)d_out;

  PW pw;
  pw.W[0]  = (const float*)d_in[3];   pw.Wb[0]  = (const float*)d_in[4];   // enc_w0/b0
  pw.W[1]  = (const float*)d_in[5];   pw.Wb[1]  = nullptr;                 // enc_w1
  pw.W[2]  = (const float*)d_in[7];   pw.Wb[2]  = nullptr;                 // enc_w2
  pw.W[3]  = (const float*)d_in[9];   pw.Wb[3]  = nullptr;                 // enc_w3
  pw.W[4]  = (const float*)d_in[11];  pw.Wb[4]  = (const float*)d_in[12];  // dec_w0/b0
  pw.W[5]  = (const float*)d_in[13];  pw.Wb[5]  = nullptr;                 // dec_w1
  pw.W[6]  = (const float*)d_in[15];  pw.Wb[6]  = nullptr;                 // dec_w2
  pw.W[7]  = (const float*)d_in[17];  pw.Wb[7]  = nullptr;                 // dec_w3
  pw.W[8]  = (const float*)d_in[19];  pw.Wb[8]  = (const float*)d_in[20];  // fc1_w/b
  pw.W[9]  = (const float*)d_in[21];  pw.Wb[9]  = nullptr;                 // fc2_w
  pw.W[10] = (const float*)d_in[23];  pw.Wb[10] = nullptr;                 // fc4_w

  const float* bn_in_g  = (const float*)d_in[25];
  const float* bn_in_b  = (const float*)d_in[26];
  const float* bn_lat_g = (const float*)d_in[27];
  const float* bn_lat_b = (const float*)d_in[28];
  const float* ep1_g = (const float*)d_in[29];
  const float* ep1_b = (const float*)d_in[30];
  const float* ep2_g = (const float*)d_in[31];
  const float* ep2_b = (const float*)d_in[32];
  const float* ep3_g = (const float*)d_in[33];
  const float* ep3_b = (const float*)d_in[34];

  const int NB_N = (NN + 255) / 256;      // 391
  const int NB_E = EE / 128;              // 12500 (exact)
  const int NB_Z = (ZCOUNT + 255) / 256;

  k_zero<<<NB_Z, 256, 0, stream>>>(ws);
  k_prep<<<1, 256, 0, stream>>>(pw, (uint32_t*)ws);

  k_xstats<<<NB_N, 256, 0, stream>>>(p);
  k_finalize<<<1, 32, 0, stream>>>(ws + SX_S, ws + SX_Q, bn_in_g, bn_in_b,
                                   ws + AX, ws + BX, 3, (float)NN);

  k_encoder<<<NB_E, 256, 0, stream>>>(p);

  k_znode<<<NB_N, 256, 0, stream>>>(p);
  k_finalize<<<1, 32, 0, stream>>>(ws + SZ_S, ws + SZ_Q, bn_lat_g, bn_lat_b,
                                   ws + AZ, ws + BZ, 2, (float)NN);

  k_zbn<<<NB_N, 256, 0, stream>>>(p);
  k_finalize<<<1, 32, 0, stream>>>(ws + SXC_S, ws + SXC_Q, ep1_g, ep1_b,
                                   ws + AXC, ws + BXC, 4, (float)EE);

  k_dec<<<NB_E, 256, 0, stream>>>(p);
  k_finalize<<<1, 32, 0, stream>>>(ws + SH_S, ws + SH_Q, ep2_g, ep2_b,
                                   ws + AH, ws + BH, 32, (float)EE);

  k_hres<<<NB_E, 256, 0, stream>>>(p);
  k_finalize<<<1, 32, 0, stream>>>(ws + SR_S, ws + SR_Q, ep3_g, ep3_b,
                                   ws + AR, ws + BR, 32, (float)EE);

  k_pred<<<NB_E, 256, 0, stream>>>(p);
  k_xrecon<<<NB_N, 256, 0, stream>>>(p);
}
